// DeformableConV2_2370821948021
// MI455X (gfx1250) — matrix operand order repaired and checked by reading, not compiled, not run on hardware
//
#include <hip/hip_runtime.h>
#include <hip/hip_bf16.h>

// DCNv2 forward for MI455X (gfx1250, wave32, WMMA).
// GEMM: out[b,o,h,w] = sum_{c,k} sampled[b,c,k,h,w] * W[o,c,k] + bias[o]
// M = B*H*W = 32768, N = O = 256, Kred = C*KK = 2304, f16 WMMA / f32 accum.
//
// v2: one workgroup covers the FULL N=256 (each wave owns two 16-wide N
// tiles sharing one A fragment -> gather done once per pixel tile), and the
// LDS A chunk is double-buffered so tap k+1's bilinear gather overlaps tap
// k's WMMAs with a single barrier per tap.

typedef __attribute__((ext_vector_type(16))) _Float16 v16h;
typedef __attribute__((ext_vector_type(8)))  float    v8f;
typedef __attribute__((ext_vector_type(4)))  float    v4f;

#define DC_B   8
#define DC_C   256
#define DC_O   256
#define DC_H   64
#define DC_W   64
#define DC_HW  4096
#define DC_KK  9
#define APITCH 264   // 256 halfs + 8 halfs pad; row = 528 B (16B aligned)

// ---------------------------------------------------------------------------
// Pre-pass: f32 weight (O,C,KK) -> f16 in WMMA-B-friendly layout:
//   wf16[ ((k*8 + cc)*256 + o)*32 + j ]  with c = cc*32 + j
// so a lane's B fragment (16 consecutive K=c values at fixed o) is one
// contiguous 32-byte region (two global_load_b128), fully L2-resident.
// ---------------------------------------------------------------------------
__global__ __launch_bounds__(256)
void dcn_convert_weight(const float* __restrict__ w, _Float16* __restrict__ wf16) {
    int i  = blockIdx.x * 256 + threadIdx.x;   // 9*8*256*32 = 589824 total
    int j  = i & 31;
    int o  = (i >> 5) & 255;
    int cc = (i >> 13) & 7;
    int k  = i >> 16;
    int c  = cc * 32 + j;
    wf16[i] = (_Float16)w[(o * DC_C + c) * DC_KK + k];
}

// ---------------------------------------------------------------------------
// Main fused gather + GEMM kernel.
// Block: 512 threads = 16 waves arranged 2(M) x 8(N), each wave computes
// two 16x16 output tiles (o and o+16). Tile per block: 32 pixels x 256 o.
// Grid: (32768/32, 1) = 1024 workgroups.
// ---------------------------------------------------------------------------
__global__ __launch_bounds__(512)
void dcn_gemm(const float* __restrict__ x,
              const float* __restrict__ coff,
              const float* __restrict__ cwm,
              const _Float16* __restrict__ wf16,
              const float* __restrict__ bias,
              float* __restrict__ out) {
    __shared__ _Float16 Alds[2][32 * APITCH];  // 2 x 16896 B (double buffer)
    __shared__ int      sIdx[288 * 4];         // 4608 B (clamped plane idx)
    __shared__ float    sWgt[288 * 4];         // 4608 B (mask-folded weights)

    const int tid  = threadIdx.x;
    const int lane = tid & 31;
    const int wave = tid >> 5;          // 0..15
    const int mi   = wave & 1;          // M half of the 32-pixel tile
    const int ni   = wave >> 1;         // 0..7 -> 32 out-channels each
    const int P0   = blockIdx.x * 32;   // global pixel base (same b, same h)
    const int o0   = ni * 32;

    // ---- per-(pixel, tap) bilinear sample precompute (288 entries) ----
    if (tid < 288) {
        int p = tid / 9, k = tid - p * 9;
        int m = P0 + p;
        int b = m >> 12;
        int s = m & 4095;
        int h = s >> 6, w = s & 63;
        int ky = k / 3, kx = k - ky * 3;
        float oy = coff[(size_t)(b * 18 + 2 * k)     * DC_HW + s];
        float ox = coff[(size_t)(b * 18 + 2 * k + 1) * DC_HW + s];
        float mk = cwm[(size_t)(b * DC_KK + k)       * DC_HW + s];
        float py = (float)(h - 1 + ky) + oy;          // stride=1, pad=1, dil=1
        float px = (float)(w - 1 + kx) + ox;
        float y0f = floorf(py), x0f = floorf(px);
        int   y0  = (int)y0f,   x0  = (int)x0f;
        float wy1 = py - y0f, wy0 = 1.0f - wy1;
        float wx1 = px - x0f, wx0 = 1.0f - wx1;
        const float wyv[2] = {wy0, wy1};
        const float wxv[2] = {wx0, wx1};
#pragma unroll
        for (int c4 = 0; c4 < 4; ++c4) {
            int dy = c4 >> 1, dx = c4 & 1;
            int yi = y0 + dy, xi = x0 + dx;
            bool valid = (yi >= 0) && (yi < DC_H) && (xi >= 0) && (xi < DC_W);
            float wgt = valid ? (wyv[dy] * wxv[dx] * mk) : 0.0f;  // zero OOB
            int yc = yi < 0 ? 0 : (yi > DC_H - 1 ? DC_H - 1 : yi);
            int xc = xi < 0 ? 0 : (xi > DC_W - 1 ? DC_W - 1 : xi);
            sIdx[tid * 4 + c4] = yc * DC_W + xc;
            sWgt[tid * 4 + c4] = wgt;
        }
    }
    __syncthreads();

    // ---- accumulators seeded with bias (C/D layout: N = lane&15) ----
    const int   on0 = o0 + (lane & 15);
    const int   on1 = on0 + 16;
    const float bv0 = bias[on0];
    const float bv1 = bias[on1];
    v8f acc0 = {bv0, bv0, bv0, bv0, bv0, bv0, bv0, bv0};
    v8f acc1 = {bv1, bv1, bv1, bv1, bv1, bv1, bv1, bv1};

    // Fragment sub-offsets (contiguous k = 16*(lane/16) + s, same order for A and B)
    const int kbaseA = (lane < 16) ? 0 : 16;   // A: K 0-15 vs 16-31
    const int jbaseB = (lane < 16) ? 0 : 16;   // B: K 0-15 vs 16-31
    const int prow   = mi * 16 + (lane & 15);  // A row (M)

    // bilinear gather of one tap's 32x256 A chunk into an LDS buffer
    auto buildA = [&](int k, _Float16* dst) {
        for (int i = tid; i < 32 * DC_C; i += 512) {
            int p = i & 31;          // lanes sweep pixels -> corner locality
            int c = i >> 5;
            int e = (p * 9 + k) * 4;
            const float* xp =
                x + (size_t)(((P0 + p) >> 12) * DC_C + c) * DC_HW;
            float v = sWgt[e + 0] * xp[sIdx[e + 0]]
                    + sWgt[e + 1] * xp[sIdx[e + 1]]
                    + sWgt[e + 2] * xp[sIdx[e + 2]]
                    + sWgt[e + 3] * xp[sIdx[e + 3]];
            dst[p * APITCH + c] = (_Float16)v;
        }
    };

    buildA(0, Alds[0]);
    __syncthreads();

    for (int k = 0; k < DC_KK; ++k) {
        // ---- overlap: gather tap k+1 into the other buffer ----
        if (k + 1 < DC_KK) buildA(k + 1, Alds[(k + 1) & 1]);

        // ---- 16 WMMAs: reduce tap k's 256 channels into 2 N tiles ----
        const _Float16* Ab = Alds[k & 1];
#pragma unroll
        for (int cc = 0; cc < 8; ++cc) {
            union { v16h h; v4f f[2]; } af, bf0, bf1;
            const v4f* ap =
                (const v4f*)&Ab[prow * APITCH + cc * 32 + kbaseA];
            af.f[0] = ap[0];     // K kbase   .. kbase+7
            af.f[1] = ap[1];     // K kbase+8 .. kbase+15
            const _Float16* bbase =
                wf16 + (size_t)((k * 8 + cc) * DC_O) * 32;
            const v4f* bp0 = (const v4f*)(bbase + (on0 * 32 + jbaseB));
            const v4f* bp1 = (const v4f*)(bbase + (on1 * 32 + jbaseB));
            bf0.f[0] = bp0[0];   // K jbase .. jbase+15 contiguous
            bf0.f[1] = bp0[1];
            bf1.f[0] = bp1[0];
            bf1.f[1] = bp1[1];
            acc0 = __builtin_amdgcn_wmma_f32_16x16x32_f16(
                false, af.h, false, bf0.h, (short)0, acc0, false, false);
            acc1 = __builtin_amdgcn_wmma_f32_16x16x32_f16(
                false, af.h, false, bf1.h, (short)0, acc1, false, false);
        }
        __syncthreads();   // next tap's buffer ready; old buffer reusable
    }

    // ---- writeback: lane covers o = on0 and on1, rows M = mrow0 + r ----
    {
        int mrow0 = (lane < 16) ? 0 : 8;
        int m0 = P0 + mi * 16 + mrow0;       // still same (b, h) row
        int b  = m0 >> 12;
        int s0 = m0 & 4095;
        float* op0 = out + (size_t)(b * DC_O + on0) * DC_HW + s0;
        float* op1 = out + (size_t)(b * DC_O + on1) * DC_HW + s0;
#pragma unroll
        for (int r = 0; r < 8; ++r) op0[r] = acc0[r];
#pragma unroll
        for (int r = 0; r < 8; ++r) op1[r] = acc1[r];
    }
}

// ---------------------------------------------------------------------------
extern "C" void kernel_launch(void* const* d_in, const int* in_sizes, int n_in,
                              void* d_out, int out_size, void* d_ws, size_t ws_size,
                              hipStream_t stream) {
    const float* x    = (const float*)d_in[0];  // (8,256,64,64)
    const float* coff = (const float*)d_in[1];  // (8,18,64,64)
    const float* cwm  = (const float*)d_in[2];  // (8,9,64,64)
    const float* wgt  = (const float*)d_in[3];  // (256,256,3,3)
    const float* bias = (const float*)d_in[4];  // (256,)
    float* out = (float*)d_out;                 // (8,256,64,64)

    _Float16* wf16 = (_Float16*)d_ws;           // needs 2304*256*2 = 1.18 MB

    // 1) weight f32 -> f16, WMMA-B layout (9*8*256*32 elements)
    dcn_convert_weight<<<2304, 256, 0, stream>>>(wgt, wf16);

    // 2) fused deformable gather + WMMA GEMM (full N per block)
    dcn_gemm<<<dim3(1024, 1), 512, 0, stream>>>(x, coff, cwm, wf16, bias, out);
}
